// MLPPredictor_23502061044548
// MI455X (gfx1250) — matrix operand order repaired and checked by reading, not compiled, not run on hardware
//
#include <hip/hip_runtime.h>

typedef __attribute__((ext_vector_type(16))) _Float16 v16h;
typedef __attribute__((ext_vector_type(8)))  float    v8f;
typedef __attribute__((ext_vector_type(4)))  _Float16 v4h;

#define DFEAT   128
#define TWO_D   256
#define TILE_M  16
#define WAVES   2    // waves per block
#define TM      4    // 16-edge tiles per wave

// ---------------------------------------------------------------------------
// Pack W1 [128 x 256] (torch [out,in], row-major f32) into per-lane WMMA
// B-fragment layout (f16). Fragment f = kc*8 + n covers B tile:
//   K = kc*32 .. +31 (input dim), N = n*16 .. +15 (output dim)
// Lane l holds 16 contiguous halfs: column N = n*16 + (l&15),
//   K = kc*32 + (l<16 ? 0 : 16) + j   (contiguous k order, same as the A loads)
// ---------------------------------------------------------------------------
__global__ void pack_w1_kernel(const float* __restrict__ W1,
                               _Float16* __restrict__ w1p) {
  int t = blockIdx.x * blockDim.x + threadIdx.x;
  if (t >= 64 * 32 * 16) return;
  int f   = t >> 9;
  int rem = t & 511;
  int l   = rem >> 4;
  int j   = rem & 15;
  int kc  = f >> 3;
  int n   = f & 7;
  int o   = n * 16 + (l & 15);
  int k   = kc * 32 + ((l < 16) ? 0 : 16) + j;
  w1p[t] = (_Float16)W1[o * TWO_D + k];
}

// Convert node features to f16 (4 elements / thread).
__global__ void cvt_x_kernel(const float* __restrict__ x,
                             _Float16* __restrict__ xh, int n4) {
  int t = blockIdx.x * blockDim.x + threadIdx.x;
  if (t >= n4) return;
  float4 v = ((const float4*)x)[t];
  v4h p = { (_Float16)v.x, (_Float16)v.y, (_Float16)v.z, (_Float16)v.w };
  ((v4h*)xh)[t] = p;
}

// ---------------------------------------------------------------------------
// Main kernel: per wave, TM 16-edge tiles. H[64x128] = A[64x256] @ W1^T via
// WMMA f16, B fragments held in registers across the 4 M-tiles per K-chunk.
// Templated on staging path so the hot variant carries no dead code.
// ---------------------------------------------------------------------------
template <int USE_F16X>
__global__ __launch_bounds__(WAVES * 32, 1)
void edge_mlp_wmma_kernel(const float* __restrict__ x,
                          const int* __restrict__ edge_index,
                          const _Float16* __restrict__ w1p,
                          const _Float16* __restrict__ xh,
                          const float* __restrict__ b1,
                          const float* __restrict__ W2,
                          const float* __restrict__ b2,
                          float* __restrict__ out,
                          int E, int ntiles) {
  __shared__ _Float16 feat[WAVES][TM * TILE_M * TWO_D];   // 32 KB per wave

  const int wave  = threadIdx.x >> 5;
  const int lane  = threadIdx.x & 31;
  const int tile0 = (blockIdx.x * WAVES + wave) * TM;
  if (tile0 >= ntiles) return;
  _Float16* fb = &feat[wave][0];

  // Lane owns endpoint (lane>>4: 0=src,1=dst) of edge (lane&15) in each tile.
  int node[TM];
  #pragma unroll
  for (int mt = 0; mt < TM; ++mt)
    node[mt] = edge_index[(lane >> 4) * E + (tile0 + mt) * TILE_M + (lane & 15)];

  // ---- Stage TM x [16 x 256] concat-gather tiles into LDS as f16 ---------
  if (USE_F16X) {
    // Pure copy from pre-converted f16 features: CDNA5 async global->LDS.
    // Each issue: 32 lanes x 16B = two 256B rows (16 lanes per row).
    const unsigned lbase = (unsigned)(uintptr_t)fb;   // low 32 bits = LDS addr
    #pragma unroll
    for (int mt = 0; mt < TM; ++mt) {
      #pragma unroll
      for (int i = 0; i < 16; ++i) {
        int rr = 2 * i + (lane >> 4);                 // row 0..31 in tile mt
        int nd = __shfl(node[mt], rr, 32);
        const _Float16* gp = xh + ((long)nd * DFEAT + (lane & 15) * 8);
        unsigned laddr = lbase +
            (((mt * TILE_M + (rr & 15)) * TWO_D +
              ((rr < 16) ? 0 : DFEAT) + (lane & 15) * 8) << 1);
        asm volatile("global_load_async_to_lds_b128 %0, %1, off"
                     :: "v"(laddr), "v"(gp) : "memory");
      }
    }
    asm volatile("s_wait_asynccnt 0x0" ::: "memory");
  } else {
    // Fallback (small d_ws): gather f32, convert, store to LDS.
    const float4* xf4 = (const float4*)x;
    #pragma unroll
    for (int mt = 0; mt < TM; ++mt) {
      #pragma unroll
      for (int r = 0; r < 32; ++r) {
        int nd = __shfl(node[mt], r, 32);
        float4 v = xf4[(long)nd * 32 + lane];
        v4h p = { (_Float16)v.x, (_Float16)v.y, (_Float16)v.z, (_Float16)v.w };
        *(v4h*)(fb + (mt * TILE_M + (r & 15)) * TWO_D +
                ((r < 16) ? 0 : DFEAT) + lane * 4) = p;
      }
    }
  }

  // ---- Second-layer constants (lane column n = t*16 + (lane&15)) ---------
  const int ncol = lane & 15;
  float w2v[8], b1v[8];
  #pragma unroll
  for (int t = 0; t < 8; ++t) {
    w2v[t] = W2[t * 16 + ncol];
    b1v[t] = b1[t * 16 + ncol];
  }

  // ---- WMMA main loop ----------------------------------------------------
  v8f c[TM][8];
  #pragma unroll
  for (int mt = 0; mt < TM; ++mt)
    #pragma unroll
    for (int n = 0; n < 8; ++n)
      c[mt][n] = (v8f){0.f,0.f,0.f,0.f,0.f,0.f,0.f,0.f};

  const int m    = lane & 15;              // A row owned by this lane
  const int koff = (lane < 16) ? 0 : 16;   // contiguous k = 16h + s (matches B pack)

  #pragma unroll
  for (int kc = 0; kc < 8; ++kc) {
    // 8 B fragments for this K-chunk, reused across TM M-tiles.
    v16h bf[8];
    #pragma unroll
    for (int n = 0; n < 8; ++n)
      bf[n] = *(const v16h*)(w1p + (((kc * 8 + n) * 32 + lane) << 4));
    #pragma unroll
    for (int mt = 0; mt < TM; ++mt) {
      const _Float16* ap = fb + (mt * TILE_M + m) * TWO_D + kc * 32 + koff;
      v16h a;
      ((uint4*)&a)[0] = *(const uint4*)(ap);
      ((uint4*)&a)[1] = *(const uint4*)(ap + 8);
      #pragma unroll
      for (int n = 0; n < 8; ++n)
        c[mt][n] = __builtin_amdgcn_wmma_f32_16x16x32_f16(
            false, a, false, bf[n], (short)0, c[mt][n], false, false);
    }
  }

  // ---- Epilogue: relu(H + b1) . w2 + b2, reduce over N across lanes ------
  const float bias2 = b2[0];
  #pragma unroll
  for (int mt = 0; mt < TM; ++mt) {
    float acc[8] = {0.f,0.f,0.f,0.f,0.f,0.f,0.f,0.f};
    #pragma unroll
    for (int t = 0; t < 8; ++t) {
      #pragma unroll
      for (int r = 0; r < 8; ++r) {
        float h = c[mt][t][r] + b1v[t];
        h = fmaxf(h, 0.0f);
        acc[r] = fmaf(h, w2v[t], acc[r]);
      }
    }
    #pragma unroll
    for (int r = 0; r < 8; ++r) {
      acc[r] += __shfl_xor(acc[r], 1, 32);
      acc[r] += __shfl_xor(acc[r], 2, 32);
      acc[r] += __shfl_xor(acc[r], 4, 32);
      acc[r] += __shfl_xor(acc[r], 8, 32);
    }
    const int eb = (tile0 + mt) * TILE_M;
    if (lane == 0) {
      #pragma unroll
      for (int r = 0; r < 8; ++r) out[eb + r] = acc[r] + bias2;       // M=0..7
    } else if (lane == 16) {
      #pragma unroll
      for (int r = 0; r < 8; ++r) out[eb + 8 + r] = acc[r] + bias2;   // M=8..15
    }
  }
}

extern "C" void kernel_launch(void* const* d_in, const int* in_sizes, int n_in,
                              void* d_out, int out_size, void* d_ws, size_t ws_size,
                              hipStream_t stream) {
  const float* x          = (const float*)d_in[0];
  const int*   edge_index = (const int*)d_in[1];
  const float* W1         = (const float*)d_in[2];
  const float* b1         = (const float*)d_in[3];
  const float* W2         = (const float*)d_in[4];
  const float* b2         = (const float*)d_in[5];
  float* out = (float*)d_out;

  const int E      = in_sizes[1] / 2;               // 1,600,000
  const int ntiles = (E + TILE_M - 1) / TILE_M;     // 100,000 (exact)
  const int nx     = in_sizes[0];                   // 100,000 * 128

  _Float16* w1p = (_Float16*)d_ws;                            // 64 KB
  _Float16* xh  = (_Float16*)((char*)d_ws + 64 * 1024);       // 25.6 MB
  const size_t need = 64 * 1024 + (size_t)nx * sizeof(_Float16);
  const int use_f16x = (ws_size >= need) ? 1 : 0;

  pack_w1_kernel<<<(64 * 32 * 16 + 255) / 256, 256, 0, stream>>>(W1, w1p);

  const int tiles_per_block = WAVES * TM;
  const int grid = (ntiles + tiles_per_block - 1) / tiles_per_block;

  if (use_f16x) {
    const int n4 = nx / 4;
    cvt_x_kernel<<<(n4 + 255) / 256, 256, 0, stream>>>(x, xh, n4);
    edge_mlp_wmma_kernel<1><<<grid, WAVES * 32, 0, stream>>>(
        x, edge_index, w1p, xh, b1, W2, b2, out, E, ntiles);
  } else {
    edge_mlp_wmma_kernel<0><<<grid, WAVES * 32, 0, stream>>>(
        x, edge_index, w1p, xh, b1, W2, b2, out, E, ntiles);
  }
}
